// DistAwareColoringTransformer_66090956751545
// MI455X (gfx1250) — matrix operand order repaired and checked by reading, not compiled, not run on hardware
//
#include <hip/hip_runtime.h>
#include <hip/hip_bf16.h>

// ---------------- problem constants ----------------
#define NT      2048
#define EDIM    512
#define HHEADS  8
#define DHEAD   64
#define LLAYERS 2

typedef __bf16 bf16_t;
typedef __attribute__((ext_vector_type(16))) bf16_t          v16bf;
typedef __attribute__((ext_vector_type(8)))  float           v8f;
typedef __attribute__((ext_vector_type(16))) unsigned short  v16u;
typedef __attribute__((ext_vector_type(8)))  unsigned short  v8u;

union V16 { v16u u; v16bf b; };

// f32 -> bf16, round-to-nearest-even
__device__ __forceinline__ unsigned short f2bf(float f) {
  unsigned int u = __float_as_uint(f);
  u += 0x7fffu + ((u >> 16) & 1u);
  return (unsigned short)(u >> 16);
}

// A-fragment (16x32 bf16, MxK). Lane row = lane%16, khalf = lane/16.
// elements 0..7  -> k = khalf*16 + 0..7       (16B contiguous)
// elements 8..15 -> k = khalf*16 + 8..15      (16B contiguous)
__device__ __forceinline__ v16bf load_a_frag(const unsigned short* p, int half) {
  v8u lo = *(const v8u*)(p + half * 16);
  v8u hi = *(const v8u*)(p + half * 16 + 8);
  V16 t;
#pragma unroll
  for (int i = 0; i < 8; ++i) { t.u[i] = lo[i]; t.u[i + 8] = hi[i]; }
  return t.b;
}

// B-fragment (32x16 bf16, KxN). Lane col = lane%16, element e -> k = (lane/16)*16 + e.
// Caller passes p pointing at the K-contiguous 16-element run: 32B contiguous.
__device__ __forceinline__ v16bf load_b_frag(const unsigned short* p) {
  V16 t; t.u = *(const v16u*)p; return t.b;
}

__device__ __forceinline__ v8f wmma_bf16(v16bf a, v16bf b, v8f c) {
  return __builtin_amdgcn_wmma_f32_16x16x32_bf16(false, a, false, b, (short)0, c,
                                                 false, false);
}

// xor-shuffle reductions over 16-lane half-groups (masks 1,2,4,8 keep halves)
__device__ __forceinline__ float xor16_max(float v) {
#pragma unroll
  for (int m = 1; m < 16; m <<= 1) {
    int o = __builtin_amdgcn_ds_bpermute((((int)(threadIdx.x & 31)) ^ m) << 2,
                                         __float_as_int(v));
    v = fmaxf(v, __int_as_float(o));
  }
  return v;
}
__device__ __forceinline__ float xor16_sum(float v) {
#pragma unroll
  for (int m = 1; m < 16; m <<= 1) {
    int o = __builtin_amdgcn_ds_bpermute((((int)(threadIdx.x & 31)) ^ m) << 2,
                                         __float_as_int(v));
    v += __int_as_float(o);
  }
  return v;
}

// ---------------- elementwise helpers ----------------
__global__ void k_f32_to_bf16(const float* __restrict__ in,
                              unsigned short* __restrict__ out, int n) {
  int i = blockIdx.x * blockDim.x + threadIdx.x;
  if (i < n) out[i] = f2bf(in[i]);
}

__global__ void k_copy_f32(const float* __restrict__ in, float* __restrict__ out, int n) {
  int i = blockIdx.x * blockDim.x + threadIdx.x;
  if (i < n) out[i] = in[i];
}

// mask[i,j] = (D<=2) ? dot(dist_emb[clamp(D,0,11)], w) + b : -1e9
__global__ void k_mask(const int* __restrict__ D, const float* __restrict__ emb,
                       const float* __restrict__ pw, const float* __restrict__ pb,
                       float* __restrict__ mask, int n) {
  int i = blockIdx.x * blockDim.x + threadIdx.x;
  if (i >= n) return;
  int d  = D[i];
  int dc = d < 0 ? 0 : (d > 11 ? 11 : d);
  float s = pb[0];
#pragma unroll
  for (int e = 0; e < 8; ++e) s += emb[dc * 8 + e] * pw[e];
  mask[i] = (d <= 2) ? s : -1e9f;
}

// ---------------- LayerNorm (one 256-thread block per row, E=512) ----------------
__device__ __forceinline__ float block_reduce_sum(float v, float* red) {
#pragma unroll
  for (int m = 1; m < 32; m <<= 1) {
    int o = __builtin_amdgcn_ds_bpermute((((int)(threadIdx.x & 31)) ^ m) << 2,
                                         __float_as_int(v));
    v += __int_as_float(o);
  }
  __syncthreads();                       // protect red[] against previous use
  if ((threadIdx.x & 31) == 0) red[threadIdx.x >> 5] = v;
  __syncthreads();
  float s = red[0];
#pragma unroll
  for (int i = 1; i < 8; ++i) s += red[i];
  return s;
}

__global__ void k_layernorm_bf16(const float* __restrict__ x,
                                 const float* __restrict__ scale,
                                 const float* __restrict__ bias,
                                 unsigned short* __restrict__ out) {
  __shared__ float red[8];
  int row = blockIdx.x, t = threadIdx.x;
  const float* xr = x + (size_t)row * EDIM;
  float a = xr[t], b = xr[t + 256];
  float mu  = block_reduce_sum(a + b, red) * (1.0f / EDIM);
  float da = a - mu, db = b - mu;
  float var = block_reduce_sum(da * da + db * db, red) * (1.0f / EDIM);
  float rs  = rsqrtf(var + 1e-5f);
  out[(size_t)row * EDIM + t]       = f2bf(da * rs * scale[t]       + bias[t]);
  out[(size_t)row * EDIM + t + 256] = f2bf(db * rs * scale[t + 256] + bias[t + 256]);
}

// ---------------- WMMA GEMM: C[m,f] = sum_k A[m,k]*W[f,k] + bias[f] ----------------
// block = 128 (4 waves); wave computes a 16x64 tile (4 accumulators, shared A-frag);
// grid = (M/16, F/256)
template <bool RELU, bool RESID, bool WBF, bool WBFT>
__global__ void k_gemm_nt(const unsigned short* __restrict__ A, int lda,
                          const unsigned short* __restrict__ W, int ldw,
                          const float* __restrict__ bias, int K,
                          float* __restrict__ resid, int ldr,
                          unsigned short* __restrict__ obf, int ldo,
                          unsigned short* __restrict__ obfT, int ldoT) {
  int wave = threadIdx.x >> 5, lane = threadIdx.x & 31;
  int n = lane & 15, half = lane >> 4;
  int m0 = blockIdx.x * 16;
  int f0 = blockIdx.y * 256 + wave * 64;
  const unsigned short* arow = A + (size_t)(m0 + n) * lda;
  const unsigned short* w0 = W + (size_t)(f0 + n) * ldw;
  const unsigned short* w1 = w0 + (size_t)16 * ldw;
  const unsigned short* w2 = w0 + (size_t)32 * ldw;
  const unsigned short* w3 = w0 + (size_t)48 * ldw;
  v8f c0 = {}, c1 = {}, c2 = {}, c3 = {};
  for (int kb = 0; kb < K; kb += 32) {
    v16bf a = load_a_frag(arow + kb, half);      // shared across the 4 col tiles
    int ko = kb + half * 16;
    v16bf b0 = load_b_frag(w0 + ko);
    v16bf b1 = load_b_frag(w1 + ko);
    v16bf b2 = load_b_frag(w2 + ko);
    v16bf b3 = load_b_frag(w3 + ko);
    // prefetch k-tile two ahead (gfx1250 global_prefetch_b8; speculative, OOB-safe)
    __builtin_prefetch(arow + kb + 64, 0, 1);
    __builtin_prefetch(w0 + ko + 64, 0, 1);
    __builtin_prefetch(w1 + ko + 64, 0, 1);
    __builtin_prefetch(w2 + ko + 64, 0, 1);
    __builtin_prefetch(w3 + ko + 64, 0, 1);
    c0 = wmma_bf16(a, b0, c0);
    c1 = wmma_bf16(a, b1, c1);
    c2 = wmma_bf16(a, b2, c2);
    c3 = wmma_bf16(a, b3, c3);
  }
  v8f acc[4] = {c0, c1, c2, c3};
#pragma unroll
  for (int tt = 0; tt < 4; ++tt) {
    int f = f0 + tt * 16 + n;
    float bv = bias[f];
    if (RESID || WBF) {
#pragma unroll
      for (int r = 0; r < 8; ++r) {
        float v = acc[tt][r] + bv;
        if (RELU) v = fmaxf(v, 0.0f);
        int m = m0 + r + 8 * half;
        if (RESID) resid[(size_t)m * ldr + f] += v;
        if (WBF)   obf[(size_t)m * ldo + f] = f2bf(v);
      }
    }
    if (WBFT) {  // transposed bf16 store: one contiguous 16B store per lane
      v8u p;
#pragma unroll
      for (int r = 0; r < 8; ++r) p[r] = f2bf(acc[tt][r] + bv);
      *(v8u*)(obfT + (size_t)f * ldoT + m0 + 8 * half) = p;
    }
  }
}

// ---------------- flash attention: wave = 16 query rows of one head ----------------
// qk: [NT,1024] bf16 (q cols 0..511, k cols 512..1023, q NOT pre-scaled)
// vT: [512, NT] bf16 (dh-major), mask: [NT,NT] f32, obf: [NT,512] bf16
__global__ void k_attention(const unsigned short* __restrict__ qk,
                            const unsigned short* __restrict__ vT,
                            const float* __restrict__ mask,
                            unsigned short* __restrict__ obf) {
  __shared__ unsigned short plds[4][16 * 32];
  int wave = threadIdx.x >> 5, lane = threadIdx.x & 31;
  int n = lane & 15, half = lane >> 4;
  int h  = blockIdx.y;
  int m0 = (blockIdx.x * 4 + wave) * 16;

  const unsigned short* qrow = qk + (size_t)(m0 + n) * 1024 + h * DHEAD;
  v16bf aq0 = load_a_frag(qrow, half);        // k = dh 0..31
  v16bf aq1 = load_a_frag(qrow + 32, half);   // k = dh 32..63

  v8f o0 = {}, o1 = {}, o2 = {}, o3 = {};
  float rmax[8], rsum[8];
#pragma unroll
  for (int r = 0; r < 8; ++r) { rmax[r] = -1e30f; rsum[r] = 0.0f; }

  for (int j0 = 0; j0 < NT; j0 += 32) {
    float s0[8], s1[8];
#pragma unroll
    for (int t = 0; t < 2; ++t) {
      int j = j0 + t * 16;
      const unsigned short* krow = qk + (size_t)(j + n) * 1024 + EDIM + h * DHEAD;
      v8f s = {};
      s = wmma_bf16(aq0, load_b_frag(krow + half * 16), s);
      s = wmma_bf16(aq1, load_b_frag(krow + 32 + half * 16), s);
      const float* mp = mask + (size_t)(m0 + 8 * half) * NT + j + n;
      __builtin_prefetch(mp + 32, 0, 1);   // next mask block
#pragma unroll
      for (int r = 0; r < 8; ++r) {
        float v = s[r] * 0.125f + mp[(size_t)r * NT];   // scale = 1/sqrt(64)
        if (t == 0) s0[r] = v; else s1[r] = v;
      }
    }
    // online softmax + stage P (bf16) in this wave's LDS tile [16 rows][32 cols]
#pragma unroll
    for (int r = 0; r < 8; ++r) {
      float tm   = xor16_max(fmaxf(s0[r], s1[r]));
      float nm   = fmaxf(rmax[r], tm);
      float corr = __expf(rmax[r] - nm);
      float p0   = __expf(s0[r] - nm);
      float p1   = __expf(s1[r] - nm);
      rsum[r] = rsum[r] * corr + xor16_sum(p0 + p1);
      rmax[r] = nm;
      o0[r] *= corr; o1[r] *= corr; o2[r] *= corr; o3[r] *= corr;
      int m = r + 8 * half;
      plds[wave][m * 32 + n]      = f2bf(p0);
      plds[wave][m * 32 + 16 + n] = f2bf(p1);
    }
    // re-layout C->A through LDS (same-wave ds ops, in order), then P x V
    v16bf pa = load_a_frag(&plds[wave][n * 32], half);
    const unsigned short* vbase = vT + (size_t)h * DHEAD * NT + j0 + half * 16;
    o0 = wmma_bf16(pa, load_b_frag(vbase + (size_t)(n)      * NT), o0);
    o1 = wmma_bf16(pa, load_b_frag(vbase + (size_t)(16 + n) * NT), o1);
    o2 = wmma_bf16(pa, load_b_frag(vbase + (size_t)(32 + n) * NT), o2);
    o3 = wmma_bf16(pa, load_b_frag(vbase + (size_t)(48 + n) * NT), o3);
  }
#pragma unroll
  for (int r = 0; r < 8; ++r) {
    float inv = 1.0f / rsum[r];
    int m = m0 + r + 8 * half;
    unsigned short* orow = obf + (size_t)m * EDIM + h * DHEAD + n;
    orow[0]  = f2bf(o0[r] * inv);
    orow[16] = f2bf(o1[r] * inv);
    orow[32] = f2bf(o2[r] * inv);
    orow[48] = f2bf(o3[r] * inv);
  }
}

// ---------------- fused color logits + softmax ----------------
__global__ void k_logits_softmax(const unsigned short* __restrict__ xbf,
                                 const unsigned short* __restrict__ cw,
                                 const float* __restrict__ cb,
                                 float* __restrict__ out) {
  int wave = threadIdx.x >> 5, lane = threadIdx.x & 31;
  int n = lane & 15, half = lane >> 4;
  int m0 = (blockIdx.x * 4 + wave) * 16;
  const unsigned short* arow = xbf + (size_t)(m0 + n) * EDIM;
  const unsigned short* wrow = cw + (size_t)n * EDIM;
  v8f c = {};
  for (int kb = 0; kb < EDIM; kb += 32)
    c = wmma_bf16(load_a_frag(arow + kb, half),
                  load_b_frag(wrow + kb + half * 16), c);
  float bv = cb[n];
#pragma unroll
  for (int r = 0; r < 8; ++r) {
    float v = c[r] + bv;
    float mx = xor16_max(v);
    float e  = __expf(v - mx);
    float s  = xor16_sum(e);
    out[(size_t)(m0 + r + 8 * half) * 16 + n] = e / s;
  }
}

extern "C" void kernel_launch(void* const* d_in, const int* in_sizes, int n_in,
                              void* d_out, int out_size, void* d_ws, size_t ws_size,
                              hipStream_t stream) {
  (void)in_sizes; (void)n_in; (void)out_size; (void)ws_size;
  const int*   D     = (const int*)  d_in[0];
  const float* x_in  = (const float*)d_in[1];
  const float* demb  = (const float*)d_in[2];
  const float* dpw   = (const float*)d_in[3];
  const float* dpb   = (const float*)d_in[4];
  const float* ln1s  = (const float*)d_in[5];
  const float* ln1b  = (const float*)d_in[6];
  const float* inw   = (const float*)d_in[7];
  const float* inb   = (const float*)d_in[8];
  const float* outw  = (const float*)d_in[9];
  const float* outb  = (const float*)d_in[10];
  const float* ln2s  = (const float*)d_in[11];
  const float* ln2b  = (const float*)d_in[12];
  const float* ff1w  = (const float*)d_in[13];
  const float* ff1b  = (const float*)d_in[14];
  const float* ff2w  = (const float*)d_in[15];
  const float* ff2b  = (const float*)d_in[16];
  const float* colw  = (const float*)d_in[17];
  const float* colb  = (const float*)d_in[18];

  char* ws = (char*)d_ws;
  size_t off = 0;
  auto alloc = [&](size_t bytes) {
    void* p = ws + off; off = (off + bytes + 255) & ~(size_t)255; return p;
  };
  float*          maskb = (float*)         alloc((size_t)NT * NT * 4);
  float*          xres  = (float*)         alloc((size_t)NT * EDIM * 4);
  unsigned short* xnbf  = (unsigned short*)alloc((size_t)NT * EDIM * 2);
  unsigned short* qkbf  = (unsigned short*)alloc((size_t)NT * 1024 * 2);
  unsigned short* vTbf  = (unsigned short*)alloc((size_t)EDIM * NT * 2);
  unsigned short* atbf  = (unsigned short*)alloc((size_t)NT * EDIM * 2);
  unsigned short* hbf   = (unsigned short*)alloc((size_t)NT * 1024 * 2);
  unsigned short* inwb  = (unsigned short*)alloc((size_t)LLAYERS * 1536 * EDIM * 2);
  unsigned short* outwb = (unsigned short*)alloc((size_t)LLAYERS * EDIM * EDIM * 2);
  unsigned short* ff1wb = (unsigned short*)alloc((size_t)LLAYERS * 1024 * EDIM * 2);
  unsigned short* ff2wb = (unsigned short*)alloc((size_t)LLAYERS * EDIM * 1024 * 2);
  unsigned short* colwb = (unsigned short*)alloc((size_t)16 * EDIM * 2);

  auto cvt = [&](const float* src, unsigned short* dst, int n) {
    k_f32_to_bf16<<<(n + 255) / 256, 256, 0, stream>>>(src, dst, n);
  };
  cvt(inw,  inwb,  LLAYERS * 1536 * EDIM);
  cvt(outw, outwb, LLAYERS * EDIM * EDIM);
  cvt(ff1w, ff1wb, LLAYERS * 1024 * EDIM);
  cvt(ff2w, ff2wb, LLAYERS * EDIM * 1024);
  cvt(colw, colwb, 16 * EDIM);

  k_copy_f32<<<(NT * EDIM + 255) / 256, 256, 0, stream>>>(x_in, xres, NT * EDIM);
  k_mask<<<(NT * NT + 255) / 256, 256, 0, stream>>>(D, demb, dpw, dpb, maskb, NT * NT);

  for (int l = 0; l < LLAYERS; ++l) {
    const unsigned short* win = inwb + (size_t)l * 1536 * EDIM;
    // --- attention block ---
    k_layernorm_bf16<<<NT, 256, 0, stream>>>(xres, ln1s + l * EDIM, ln1b + l * EDIM, xnbf);
    // q|k projection (row-major bf16), F = 1024
    k_gemm_nt<false, false, true, false><<<dim3(NT / 16, 1024 / 256), 128, 0, stream>>>(
        xnbf, EDIM, win, EDIM, inb + (size_t)l * 1536, EDIM,
        nullptr, 0, qkbf, 1024, nullptr, 0);
    // v projection -> transposed bf16 [512, NT], F = 512
    k_gemm_nt<false, false, false, true><<<dim3(NT / 16, EDIM / 256), 128, 0, stream>>>(
        xnbf, EDIM, win + (size_t)1024 * EDIM, EDIM, inb + (size_t)l * 1536 + 1024, EDIM,
        nullptr, 0, nullptr, 0, vTbf, NT);
    k_attention<<<dim3(NT / 64, HHEADS), 128, 0, stream>>>(qkbf, vTbf, maskb, atbf);
    // out projection + residual into xres
    k_gemm_nt<false, true, false, false><<<dim3(NT / 16, EDIM / 256), 128, 0, stream>>>(
        atbf, EDIM, outwb + (size_t)l * EDIM * EDIM, EDIM, outb + (size_t)l * EDIM, EDIM,
        xres, EDIM, nullptr, 0, nullptr, 0);
    // --- FFN block ---
    k_layernorm_bf16<<<NT, 256, 0, stream>>>(xres, ln2s + l * EDIM, ln2b + l * EDIM, xnbf);
    k_gemm_nt<true, false, true, false><<<dim3(NT / 16, 1024 / 256), 128, 0, stream>>>(
        xnbf, EDIM, ff1wb + (size_t)l * 1024 * EDIM, EDIM, ff1b + (size_t)l * 1024, EDIM,
        nullptr, 0, hbf, 1024, nullptr, 0);
    k_gemm_nt<false, true, false, false><<<dim3(NT / 16, EDIM / 256), 128, 0, stream>>>(
        hbf, 1024, ff2wb + (size_t)l * EDIM * 1024, 1024, ff2b + (size_t)l * EDIM, 1024,
        xres, EDIM, nullptr, 0, nullptr, 0);
  }

  k_f32_to_bf16<<<(NT * EDIM + 255) / 256, 256, 0, stream>>>(xres, xnbf, NT * EDIM);
  k_logits_softmax<<<NT / 64, 128, 0, stream>>>(xnbf, colwb, colb, (float*)d_out);
}
